// EulerAttentionHead_75892072121105
// MI455X (gfx1250) — matrix operand order repaired and checked by reading, not compiled, not run on hardware
//
#include <hip/hip_runtime.h>
#include <math.h>

typedef _Float16 half_t;
typedef __attribute__((ext_vector_type(16))) _Float16 v16h;
typedef __attribute__((ext_vector_type(8)))  _Float16 v8h;
typedef __attribute__((ext_vector_type(8)))  float    v8f;

#define BATCHN 4
#define SEQN   4096
#define DMODEL 1024
#define HDIM   128
#define FDIM   256   // 2*HDIM: [cos || sin]

// Attention LDS layout (dynamic smem): double-buffered K^T and V tiles + P buf
#define CK_STRIDE 40                        // halves per feature row (padded)
#define CK_BYTES  (FDIM * CK_STRIDE * 2)    // 20480 B
#define V_BYTES   (32 * HDIM * 2)           // 8192 B
#define OFF_CK0   0
#define OFF_CK1   (CK_BYTES)
#define OFF_V0    (2 * CK_BYTES)
#define OFF_V1    (2 * CK_BYTES + V_BYTES)
#define OFF_PSH   (2 * CK_BYTES + 2 * V_BYTES)
#define SMEM_ATTN (OFF_PSH + 4 * 16 * 32 * 2)   // 61440 B

__device__ __forceinline__ v16h mk16(v8h lo, v8h hi) {
  return __builtin_shufflevector(lo, hi, 0,1,2,3,4,5,6,7,8,9,10,11,12,13,14,15);
}
__device__ __forceinline__ v8f wmma_f16(v16h a, v16h b, v8f c) {
  return __builtin_amdgcn_wmma_f32_16x16x32_f16(false, a, false, b, (short)0, c,
                                                false, false);
}
__device__ __forceinline__ v8f vzero8() {
  v8f z;
#pragma unroll
  for (int i = 0; i < 8; ++i) z[i] = 0.0f;
  return z;
}

// ---------------------------------------------------------------------------
// Tensor Data Mover (gfx1250). Guarded: falls back to manual copies if the
// builtin is absent. 5-arg form on ROCm clang-22, 6-arg on clang-23+.
// ---------------------------------------------------------------------------
#if __has_builtin(__builtin_amdgcn_tensor_load_to_lds) && \
    __has_builtin(__builtin_amdgcn_s_wait_tensorcnt)
#define HAVE_TDM 1
typedef __attribute__((ext_vector_type(4))) unsigned int u32x4;
typedef __attribute__((ext_vector_type(8))) int i32x8;
typedef __attribute__((ext_vector_type(4))) int i32x4;

// Pack loop-invariant D# group-1: 2-byte elements, 2-D tile (tile_d0 x tile_d1)
// from global rows of stride0 elements; optional LDS padding control in e0.
#define TDM_G1(e0_extra, tensor_d0, tensor_d1, tile_d0, tile_d1, stride0)      \
  { (int)((1u << 16) | (e0_extra)),                                            \
    (int)(((tensor_d0) & 0xFFFFu) << 16),                                      \
    (int)((((tensor_d0) >> 16) & 0xFFFFu) | (((tensor_d1) & 0xFFFFu) << 16)),  \
    (int)((((tensor_d1) >> 16) & 0xFFFFu) | (((tile_d0) & 0xFFFFu) << 16)),    \
    (int)((tile_d1) & 0xFFFFu),                                                \
    (int)((unsigned)(stride0)), 0, 0 }

__device__ __forceinline__ void tdm_issue(unsigned long long ga,
                                          unsigned lds_byte_off,
                                          const i32x8& g1)
{
  u32x4 g0;
  g0[0] = 1u;                                          // count=1, user D#
  g0[1] = lds_byte_off;                                // lds_addr
  g0[2] = (unsigned)ga;                                // global_addr[31:0]
  g0[3] = (unsigned)((ga >> 32) & 0x01FFFFFFu) | (2u << 30);  // addr[56:32]|type=2
  const i32x4 z4 = {0, 0, 0, 0};
#if __clang_major__ >= 23
  const i32x8 z8 = {0, 0, 0, 0, 0, 0, 0, 0};
  __builtin_amdgcn_tensor_load_to_lds(g0, g1, z4, z4, z8, 0);
#else
  __builtin_amdgcn_tensor_load_to_lds(g0, g1, z4, z4, 0);
#endif
}
#endif

// ---------------------------------------------------------------------------
// Kernel 1: C = x @ W^T + b.  which==0: Q -> [cos||sin] f16, row-major [row][256]
//           which==1: K -> [cos||sin] f16, feature-major [b][256][4096]
//           which==2: V -> f16 row-major [row][128]
// ---------------------------------------------------------------------------
__global__ __launch_bounds__(128) void qkv_proj_kernel(
    const float* __restrict__ x, const float* __restrict__ wavelengths,
    const float* __restrict__ phase_bias, const float* __restrict__ W,
    const float* __restrict__ bias, int which, half_t* __restrict__ out)
{
  __shared__ alignas(16) half_t xs[64 * 32];   // [row][k]  row stride 32
  __shared__ alignas(16) half_t wT[32 * 128];  // [k][n]    row stride 128
  const int tid = threadIdx.x, lane = tid & 31, wv = tid >> 5;
  const int rowbase = blockIdx.x * 64;
  const int r = lane & 15, hi = lane >> 4;

  v8f acc[8];
#pragma unroll
  for (int i = 0; i < 8; ++i) acc[i] = vzero8();

  for (int k0 = 0; k0 < DMODEL; k0 += 32) {
    { // stage 64x32 of x, f32 -> f16
      const int rr = tid >> 1, hr = tid & 1;
      const float* src = x + (size_t)(rowbase + rr) * DMODEL + k0 + hr * 16;
      const float4 f0 = ((const float4*)src)[0];
      const float4 f1 = ((const float4*)src)[1];
      const float4 f2 = ((const float4*)src)[2];
      const float4 f3 = ((const float4*)src)[3];
      v8h a, b;
      a[0]=(half_t)f0.x; a[1]=(half_t)f0.y; a[2]=(half_t)f0.z; a[3]=(half_t)f0.w;
      a[4]=(half_t)f1.x; a[5]=(half_t)f1.y; a[6]=(half_t)f1.z; a[7]=(half_t)f1.w;
      b[0]=(half_t)f2.x; b[1]=(half_t)f2.y; b[2]=(half_t)f2.z; b[3]=(half_t)f2.w;
      b[4]=(half_t)f3.x; b[5]=(half_t)f3.y; b[6]=(half_t)f3.z; b[7]=(half_t)f3.w;
      *(v8h*)&xs[rr * 32 + hr * 16]     = a;
      *(v8h*)&xs[rr * 32 + hr * 16 + 8] = b;
    }
    { // stage W^T: thread tid owns output column n=tid; scatter k into rows
      const float* wsrc = W + (size_t)tid * DMODEL + k0;
#pragma unroll
      for (int j = 0; j < 32; j += 4) {
        const float4 f = *(const float4*)(wsrc + j);
        wT[(j + 0) * 128 + tid] = (half_t)f.x;
        wT[(j + 1) * 128 + tid] = (half_t)f.y;
        wT[(j + 2) * 128 + tid] = (half_t)f.z;
        wT[(j + 3) * 128 + tid] = (half_t)f.w;
      }
    }
    __syncthreads();
    const half_t* ap = &xs[(wv * 16 + r) * 32 + hi * 8];
    const v16h a = mk16(*(const v8h*)ap, *(const v8h*)(ap + 16));
#pragma unroll
    for (int nt = 0; nt < 8; ++nt) {
      const half_t* bp = &wT[(hi * 8) * 128 + nt * 16 + r];
      v16h bb;
#pragma unroll
      for (int s = 0; s < 8; ++s) { bb[s] = bp[s * 128]; bb[s + 8] = bp[(s + 16) * 128]; }
      acc[nt] = wmma_f16(a, bb, acc[nt]);
    }
    __syncthreads();
  }

  // epilogue (C layout: reg ri -> row mbase+ri, lane&15 -> col)
  const int mbase = (lane < 16) ? 0 : 8;
  const int coll = lane & 15;
  const int grow = rowbase + wv * 16 + mbase;
#pragma unroll
  for (int nt = 0; nt < 8; ++nt) {
    const int col = nt * 16 + coll;
    const float bc = bias[col];
    if (which < 2) {
      const float iw = 1.0f / (wavelengths[col] + 1e-8f);
      const float pb = phase_bias[col];
#pragma unroll
      for (int ri = 0; ri < 8; ++ri) {
        const int row = grow + ri;
        const float th = (acc[nt][ri] + bc) * iw + pb;
        const half_t c = (half_t)__cosf(th);
        const half_t s = (half_t)__sinf(th);
        if (which == 0) {                       // Q: row-major [row][256]
          out[(size_t)row * FDIM + col]        = c;
          out[(size_t)row * FDIM + HDIM + col] = s;
        } else {                                // K: feature-major [b][256][S]
          const int bb = row >> 12, ss = row & (SEQN - 1);
          out[((size_t)bb * FDIM + col) * SEQN + ss]        = c;
          out[((size_t)bb * FDIM + HDIM + col) * SEQN + ss] = s;
        }
      }
    } else {
#pragma unroll
      for (int ri = 0; ri < 8; ++ri) {
        const int row = grow + ri;
        out[(size_t)row * HDIM + col] = (half_t)(acc[nt][ri] + bc);
      }
    }
  }
}

// ---------------------------------------------------------------------------
// Kernel 2: flash attention. 128 thr (4 waves), 64 query rows/block, key tiles
// of 32. K^T / V tiles DMA'd by the TDM into double-buffered LDS.
// ---------------------------------------------------------------------------
__global__ __launch_bounds__(128) void attn_kernel(
    const half_t* __restrict__ Cq, const half_t* __restrict__ CkT,
    const half_t* __restrict__ Vh, half_t* __restrict__ Oh)
{
  extern __shared__ __align__(16) char smem[];
  const int tid = threadIdx.x, lane = tid & 31, wv = tid >> 5;
  const int b = blockIdx.y;
  const int q0 = blockIdx.x * 64 + wv * 16;
  const size_t cqb = (size_t)b * SEQN * FDIM;
  const size_t ckb = (size_t)b * FDIM * SEQN;
  const size_t vbb = (size_t)b * SEQN * HDIM;
  const int r = lane & 15, hi = lane >> 4;

  half_t* pw = (half_t*)(smem + OFF_PSH) + wv * (16 * 32);

  // Q fragments (A layout): 8 k-steps of 32 features, resident in VGPRs
  v16h aq[8];
  {
    const half_t* qrow = Cq + cqb + (size_t)(q0 + r) * FDIM + hi * 8;
#pragma unroll
    for (int ks = 0; ks < 8; ++ks) {
      const half_t* p = qrow + ks * 32;
      aq[ks] = mk16(*(const v8h*)p, *(const v8h*)(p + 16));
    }
  }

  v8f o[8];
#pragma unroll
  for (int i = 0; i < 8; ++i) o[i] = vzero8();
  float mrow[8], lrow[8];
#pragma unroll
  for (int i = 0; i < 8; ++i) { mrow[i] = -1e30f; lrow[i] = 0.0f; }
  const float scale = 0.088388347648318447f;  // 1/sqrt(128)
  const float l2e   = 1.4426950408889634f;

  // Provably uniform leader flag -> scalar branch around TDM issue (tensor
  // instructions execute regardless of EXEC, so a real branch is required).
  const bool leader = (__builtin_amdgcn_readfirstlane(tid) < 32);

#ifdef HAVE_TDM
  // Loop-invariant descriptor group-1 words.
  // K^T tile: 256 rows x 32 halves, row stride SEQN halves; pad 4 DWORDs after
  // every 16 DWORDs -> LDS row stride 40 halves.
  const i32x8 g1_ck = TDM_G1((1u << 20) | (3u << 22) | (3u << 25),
                             SEQN, FDIM, 32, FDIM, SEQN);
  // V tile: contiguous 32*128 halves (1-D).
  const i32x8 g1_v  = TDM_G1(0u, 32 * HDIM, 1, 32 * HDIM, 1, 32 * HDIM);
  const unsigned long long ckA = (unsigned long long)(CkT + ckb);
  const unsigned long long vA  = (unsigned long long)(Vh + vbb);
#endif

  auto stage = [&](int bufi, int kt) {
#ifdef HAVE_TDM
    if (leader) {
      tdm_issue(ckA + (unsigned long long)kt * 2u,
                (unsigned)(bufi ? OFF_CK1 : OFF_CK0), g1_ck);
      tdm_issue(vA + (unsigned long long)kt * (HDIM * 2u),
                (unsigned)(bufi ? OFF_V1 : OFF_V0), g1_v);
    }
#else
    {
      half_t* ck = (half_t*)(smem + (bufi ? OFF_CK1 : OFF_CK0));
      const half_t* src = CkT + ckb + kt;
#pragma unroll
      for (int rr = 0; rr < 2; ++rr) {
        const int f = tid + rr * 128;
        const v8h* s = (const v8h*)(src + (size_t)f * SEQN);
        v8h* d = (v8h*)&ck[f * CK_STRIDE];
        d[0] = s[0]; d[1] = s[1]; d[2] = s[2]; d[3] = s[3];
      }
      half_t* vd = (half_t*)(smem + (bufi ? OFF_V1 : OFF_V0));
      const half_t* vs = Vh + vbb + (size_t)kt * HDIM;
#pragma unroll
      for (int it = 0; it < 4; ++it) {
        const int off = (tid + it * 128) * 8;
        *(v8h*)&vd[off] = *(const v8h*)(vs + off);
      }
    }
#endif
  };

  stage(0, 0);
#ifdef HAVE_TDM
  if (leader) __builtin_amdgcn_s_wait_tensorcnt(0);
#endif
  __syncthreads();

  for (int kt = 0; kt < SEQN; kt += 32) {
    const int cur = (kt >> 5) & 1;
    if (kt + 32 < SEQN) stage(cur ^ 1, kt + 32);   // overlap DMA with compute
    const half_t* ck  = (const half_t*)(smem + (cur ? OFF_CK1 : OFF_CK0));
    const half_t* vsh = (const half_t*)(smem + (cur ? OFF_V1 : OFF_V0));

    // S = Cq * Ck^T for 32 keys: two 16-wide column tiles
    v8f s0 = vzero8(), s1 = vzero8();
#pragma unroll
    for (int ks = 0; ks < 8; ++ks) {
      const half_t* bp = &ck[(ks * 32 + hi * 8) * CK_STRIDE + r];  // lane&15 = key, elements = features
      v16h b0, b1;
#pragma unroll
      for (int s = 0; s < 8; ++s) {
        b0[s] = bp[s * CK_STRIDE];           b0[s + 8] = bp[(s + 16) * CK_STRIDE];
        b1[s] = bp[s * CK_STRIDE + 16];      b1[s + 8] = bp[(s + 16) * CK_STRIDE + 16];
      }
      s0 = wmma_f16(aq[ks], b0, s0);
      s1 = wmma_f16(aq[ks], b1, s1);
    }

    // online softmax (rows live in 16-lane halves; reduce with xor shuffles)
    float alpha[8];
#pragma unroll
    for (int ri = 0; ri < 8; ++ri) {
      const float v0 = s0[ri] * scale, v1 = s1[ri] * scale;
      float mx = fmaxf(v0, v1);
#pragma unroll
      for (int d = 8; d >= 1; d >>= 1) mx = fmaxf(mx, __shfl_xor(mx, d, 32));
      const float mn = fmaxf(mrow[ri], mx);
      alpha[ri] = exp2f((mrow[ri] - mn) * l2e);
      mrow[ri] = mn;
      const float p0 = exp2f((v0 - mn) * l2e);
      const float p1 = exp2f((v1 - mn) * l2e);
      s0[ri] = p0; s1[ri] = p1;
      float sum = p0 + p1;
#pragma unroll
      for (int d = 8; d >= 1; d >>= 1) sum += __shfl_xor(sum, d, 32);
      lrow[ri] = lrow[ri] * alpha[ri] + sum;
    }
#pragma unroll
    for (int nt = 0; nt < 8; ++nt)
#pragma unroll
      for (int ri = 0; ri < 8; ++ri) o[nt][ri] *= alpha[ri];

    // C-layout P -> LDS -> A-layout fragment (same-wave DS ops stay ordered)
    const int mb = (lane < 16) ? 0 : 8;
#pragma unroll
    for (int ri = 0; ri < 8; ++ri) {
      pw[(mb + ri) * 32 + (lane & 15)]      = (half_t)s0[ri];
      pw[(mb + ri) * 32 + 16 + (lane & 15)] = (half_t)s1[ri];
    }
    const half_t* pr = &pw[r * 32 + hi * 8];
    const v16h pa = mk16(*(const v8h*)pr, *(const v8h*)(pr + 16));

    // O += P @ V  (V gathered column-wise into B layout)
#pragma unroll
    for (int nt = 0; nt < 8; ++nt) {
      const half_t* vp = &vsh[(hi * 8) * HDIM + nt * 16 + r];
      v16h bv;
#pragma unroll
      for (int s = 0; s < 8; ++s) { bv[s] = vp[s * HDIM]; bv[s + 8] = vp[(s + 16) * HDIM]; }
      o[nt] = wmma_f16(pa, bv, o[nt]);
    }
#ifdef HAVE_TDM
    if (leader) __builtin_amdgcn_s_wait_tensorcnt(0);
#endif
    __syncthreads();
  }

  // finalize: O /= l, write f16
  float invl[8];
#pragma unroll
  for (int ri = 0; ri < 8; ++ri) invl[ri] = 1.0f / lrow[ri];
  const int mb = (lane < 16) ? 0 : 8;
#pragma unroll
  for (int nt = 0; nt < 8; ++nt) {
    const int col = nt * 16 + (lane & 15);
#pragma unroll
    for (int ri = 0; ri < 8; ++ri) {
      Oh[vbb + (size_t)(q0 + mb + ri) * HDIM + col] = (half_t)(o[nt][ri] * invl[ri]);
    }
  }
}

// ---------------------------------------------------------------------------
// Kernel 3: y = Oh @ Wo^T + bo  (M=16384, N=1024, K=128), f32 out.
// ---------------------------------------------------------------------------
__global__ __launch_bounds__(128) void out_proj_kernel(
    const half_t* __restrict__ Oh, const float* __restrict__ Wo,
    const float* __restrict__ bo, float* __restrict__ y)
{
  __shared__ alignas(16) half_t osh[64 * 32];   // [row][k]
  __shared__ alignas(16) half_t woT[32 * 128];  // [k][n]
  const int tid = threadIdx.x, lane = tid & 31, wv = tid >> 5;
  const int rowbase = blockIdx.x * 64;
  const int nbase = blockIdx.y * 128;
  const int r = lane & 15, hi = lane >> 4;

  v8f acc[8];
#pragma unroll
  for (int i = 0; i < 8; ++i) acc[i] = vzero8();

  for (int k0 = 0; k0 < HDIM; k0 += 32) {
    {
      const int rr = tid >> 1, hr = tid & 1;
      const half_t* src = Oh + (size_t)(rowbase + rr) * HDIM + k0 + hr * 16;
      *(v8h*)&osh[rr * 32 + hr * 16]     = *(const v8h*)src;
      *(v8h*)&osh[rr * 32 + hr * 16 + 8] = *(const v8h*)(src + 8);
    }
    {
      const float* wsrc = Wo + (size_t)(nbase + tid) * HDIM + k0;
#pragma unroll
      for (int j = 0; j < 32; j += 4) {
        const float4 f = *(const float4*)(wsrc + j);
        woT[(j + 0) * 128 + tid] = (half_t)f.x;
        woT[(j + 1) * 128 + tid] = (half_t)f.y;
        woT[(j + 2) * 128 + tid] = (half_t)f.z;
        woT[(j + 3) * 128 + tid] = (half_t)f.w;
      }
    }
    __syncthreads();
    const half_t* ap = &osh[(wv * 16 + r) * 32 + hi * 8];
    const v16h a = mk16(*(const v8h*)ap, *(const v8h*)(ap + 16));
#pragma unroll
    for (int nt = 0; nt < 8; ++nt) {
      const half_t* bp = &woT[(hi * 8) * 128 + nt * 16 + r];
      v16h bb;
#pragma unroll
      for (int s = 0; s < 8; ++s) { bb[s] = bp[s * 128]; bb[s + 8] = bp[(s + 16) * 128]; }
      acc[nt] = wmma_f16(a, bb, acc[nt]);
    }
    __syncthreads();
  }

  const int mb = (lane < 16) ? 0 : 8;
  const int coll = lane & 15;
#pragma unroll
  for (int nt = 0; nt < 8; ++nt) {
    const int col = nbase + nt * 16 + coll;
    const float bc = bo[col];
#pragma unroll
    for (int ri = 0; ri < 8; ++ri) {
      const size_t row = (size_t)(rowbase + wv * 16 + mb + ri);
      y[row * DMODEL + col] = acc[nt][ri] + bc;
    }
  }
}

// ---------------------------------------------------------------------------
extern "C" void kernel_launch(void* const* d_in, const int* in_sizes, int n_in,
                              void* d_out, int out_size, void* d_ws, size_t ws_size,
                              hipStream_t stream) {
  (void)in_sizes; (void)n_in; (void)out_size; (void)ws_size;
  const float* x  = (const float*)d_in[0];
  const float* wl = (const float*)d_in[1];
  const float* pb = (const float*)d_in[2];
  const float* Wq = (const float*)d_in[3];
  const float* bq = (const float*)d_in[4];
  const float* Wk = (const float*)d_in[5];
  const float* bk = (const float*)d_in[6];
  const float* Wv = (const float*)d_in[7];
  const float* bv = (const float*)d_in[8];
  const float* Wo = (const float*)d_in[9];
  const float* bo = (const float*)d_in[10];
  float* y = (float*)d_out;

  // workspace: Cq (8MB) | CkT (8MB, feature-major) | Vh (4MB) | Oh (4MB), f16
  half_t* Cq  = (half_t*)d_ws;
  half_t* CkT = Cq  + (size_t)BATCHN * SEQN * FDIM;
  half_t* Vh  = CkT + (size_t)BATCHN * SEQN * FDIM;
  half_t* Oh  = Vh  + (size_t)BATCHN * SEQN * HDIM;

  const int Mtiles = (BATCHN * SEQN) / 64;  // 256
  qkv_proj_kernel<<<dim3(Mtiles), 128, 0, stream>>>(x, wl, pb, Wq, bq, 0, Cq);
  qkv_proj_kernel<<<dim3(Mtiles), 128, 0, stream>>>(x, wl, pb, Wk, bk, 1, CkT);
  qkv_proj_kernel<<<dim3(Mtiles), 128, 0, stream>>>(x, wl, pb, Wv, bv, 2, Vh);
  attn_kernel<<<dim3(SEQN / 64, BATCHN), 128, SMEM_ATTN, stream>>>(Cq, CkT, Vh, Oh);
  out_proj_kernel<<<dim3(Mtiles, DMODEL / 128), 128, 0, stream>>>(Oh, Wo, bo, y);
}
